// NextLineModel_35278861369478
// MI455X (gfx1250) — matrix operand order repaired and checked by reading, not compiled, not run on hardware
//
#include <hip/hip_runtime.h>
#include <hip/hip_bf16.h>
#include <stdint.h>
#include <math.h>

typedef __attribute__((ext_vector_type(16))) _Float16 v16h;
typedef __attribute__((ext_vector_type(8)))  float    v8f;

#define B_SZ 256
#define S_SZ 30
#define TDEC 29
#define V_SZ 32000
#define R_SZ 512
#define E_SZ 128
#define RE_SZ 32
#define H_SZ 256
#define G4H  1024   // 4*H

// ---------------------------------------------------------------------------
// WMMA fragment helpers (CDNA5 wave32 layouts, ISA 7.12.2)
// ---------------------------------------------------------------------------
__device__ __forceinline__ v8f wmma_f16(v16h a, v16h b, v8f c) {
  // (neg_a, A, neg_b, B, c_mod, C, reuse_a, reuse_b)
  return __builtin_amdgcn_wmma_f32_16x16x32_f16(false, a, false, b, (short)0, c,
                                                false, false);
}

// A fragment (16x32 f16, MxK) from a 16-row x ldk LDS tile, k-tile = kt.
// lane 0-15 : row=lane,    K in {0..7} (v0-3) and {16..23} (v4-7)
// lane16-31 : row=lane-16, K in {8..15} and {24..31}
__device__ __forceinline__ v16h load_a_frag(const _Float16* sA, int ldk, int kt,
                                            int lane) {
  union { v16h v; uint32_t u[8]; } a;
  int row = lane & 15, hi = lane >> 4;
  const uint32_t* p32 = (const uint32_t*)(sA + row * ldk + kt * 32);
#pragma unroll
  for (int j = 0; j < 4; ++j) a.u[j] = p32[4 * hi + j];         // k = 8*hi+2j
#pragma unroll
  for (int j = 0; j < 4; ++j) a.u[4 + j] = p32[8 + 4 * hi + j]; // k = 16+8*hi+2j
  return a.v;
}

// B fragment (32x16 f16, KxN) from pre-packed fragment-major memory:
// layout index = ((nt*KT + kt)*32 + lane)*16 + e  -> 32 contiguous bytes/lane
__device__ __forceinline__ v16h load_b_frag(const _Float16* Bpk, int nt, int kt,
                                            int KT, int lane) {
  const uint4* p = (const uint4*)(Bpk + ((size_t)(nt * KT + kt) * 32 + lane) * 16);
  union { v16h v; uint4 q[2]; } b;
  b.q[0] = p[0];
  b.q[1] = p[1];
  return b.v;
}

// ---------------------------------------------------------------------------
// Pack a row-major f32 weight W[Ng][Kg] (used as B = W^T) into f16 fragments.
// ---------------------------------------------------------------------------
__global__ void pack_b_kernel(const float* __restrict__ W,
                              _Float16* __restrict__ dst, int Ng, int Kg) {
  int KT = Kg >> 5;
  long long total = (long long)Ng * Kg;
  long long d = (long long)blockIdx.x * blockDim.x + threadIdx.x;
  if (d >= total) return;
  int e = (int)(d & 15);
  int lane = (int)((d >> 4) & 31);
  int kt = (int)((d >> 9) % KT);
  int nt = (int)(d / (512LL * KT));
  int n = nt * 16 + (lane & 15);
  int k = kt * 32 + ((e >> 3) << 4) + ((lane >> 4) << 3) + (e & 7);
  dst[d] = (_Float16)W[(size_t)n * Kg + k];
}

// Gather token embeddings to f16 rows (row r = b*T_eff + t). PAD(=0) -> 0.
__global__ void gather_emb_kernel(const float* __restrict__ W_emb,
                                  const int* __restrict__ seq,
                                  _Float16* __restrict__ dst, int rows,
                                  int T_eff, int T_src) {
  int idx = blockIdx.x * blockDim.x + threadIdx.x;
  if (idx >= rows * E_SZ) return;
  int r = idx / E_SZ, e = idx % E_SZ;
  int b = r / T_eff, t = r % T_eff;
  int tok = seq[b * T_src + t];
  float v = (tok == 0) ? 0.f : W_emb[(size_t)tok * E_SZ + e];
  dst[idx] = (_Float16)v;
}

__global__ void init_state_kernel(float* __restrict__ h, float* __restrict__ c,
                                  _Float16* __restrict__ h16) {
  int i = blockIdx.x * blockDim.x + threadIdx.x;
  if (i >= B_SZ * H_SZ) return;
  h[i] = 0.f;
  c[i] = 0.f;
  h16[i] = (_Float16)0.f;
}

// ---------------------------------------------------------------------------
// xproj = emb @ Wih^T + (bih+bhh), stored time-major [t][b][4H].
// Block = 16 A-rows x 1024 cols; 8 waves x 8 n-tiles; K = 128 (4 k-tiles).
// A fragments hoisted to registers (invariant across n-tiles).
// ---------------------------------------------------------------------------
__global__ __launch_bounds__(256) void gemm_xproj_kernel(
    const _Float16* __restrict__ A, const _Float16* __restrict__ Bpk,
    const float* __restrict__ bih, const float* __restrict__ bhh,
    float* __restrict__ xproj, int T) {
  __shared__ _Float16 sA[16 * E_SZ];
  int tid = threadIdx.x;
  int m0 = blockIdx.x * 16;
  const uint32_t* Ag = (const uint32_t*)(A + (size_t)m0 * E_SZ);
  uint32_t* sAu = (uint32_t*)sA;
  for (int i = tid; i < 16 * E_SZ / 2; i += 256) sAu[i] = Ag[i];
  __syncthreads();

  int lane = tid & 31, w = tid >> 5;
  int hi = lane >> 4, ln = lane & 15;

  v16h afr[4];
#pragma unroll
  for (int kt = 0; kt < 4; ++kt) afr[kt] = load_a_frag(sA, E_SZ, kt, lane);

  for (int nti = 0; nti < 8; ++nti) {
    int nt = w * 8 + nti;
    int col = nt * 16 + ln;
    v8f acc = {0.f, 0.f, 0.f, 0.f, 0.f, 0.f, 0.f, 0.f};
#pragma unroll
    for (int kt = 0; kt < 4; ++kt) {
      v16h b = load_b_frag(Bpk, nt, kt, 4, lane);
      acc = wmma_f16(afr[kt], b, acc);
    }
    float bb = bih[col] + bhh[col];
#pragma unroll
    for (int r = 0; r < 8; ++r) {
      int rg = m0 + r + 8 * hi;
      int bi = rg / T, s = rg % T;
      xproj[(size_t)(s * B_SZ + bi) * G4H + col] = acc[r] + bb;
    }
  }
}

// ---------------------------------------------------------------------------
// One LSTM time step: gates = xs + h16 @ Whh^T ; cell update in place.
// 16 blocks (one per 16 batch rows). 8 waves x 8 n-tiles; K=256 (8 k-tiles).
// A fragments hoisted to registers; activated gates exchanged via LDS (f16).
// ---------------------------------------------------------------------------
__global__ __launch_bounds__(256) void lstm_step_kernel(
    const float* __restrict__ xs, const _Float16* __restrict__ WhhPk,
    _Float16* __restrict__ h16, float* __restrict__ h, float* __restrict__ c,
    _Float16* __restrict__ hs16, int t) {
  __shared__ _Float16 sA[16 * H_SZ];   // 8 KB
  __shared__ _Float16 sG[16 * G4H];    // 32 KB (activated gates)
  int tid = threadIdx.x;
  int b0 = blockIdx.x * 16;
  const uint32_t* Ag = (const uint32_t*)(h16 + (size_t)b0 * H_SZ);
  uint32_t* sAu = (uint32_t*)sA;
  for (int i = tid; i < 16 * H_SZ / 2; i += 256) sAu[i] = Ag[i];
  __syncthreads();

  int lane = tid & 31, w = tid >> 5;
  int hi = lane >> 4, ln = lane & 15;

  v16h afr[8];
#pragma unroll
  for (int kt = 0; kt < 8; ++kt) afr[kt] = load_a_frag(sA, H_SZ, kt, lane);

  for (int nti = 0; nti < 8; ++nti) {
    int nt = w * 8 + nti;
    int col = nt * 16 + ln;
    v8f acc = {0.f, 0.f, 0.f, 0.f, 0.f, 0.f, 0.f, 0.f};
#pragma unroll
    for (int kt = 0; kt < 8; ++kt) {
      v16h b = load_b_frag(WhhPk, nt, kt, 8, lane);
      acc = wmma_f16(afr[kt], b, acc);
    }
    int gid = nt >> 4;  // 0=i 1=f 2=g 3=o (tiles never straddle gate blocks)
#pragma unroll
    for (int r = 0; r < 8; ++r) {
      float g = acc[r] + xs[(size_t)(b0 + r + 8 * hi) * G4H + col];
      float act = (gid == 2) ? tanhf(g) : 1.f / (1.f + expf(-g));
      sG[(r + 8 * hi) * G4H + col] = (_Float16)act;
    }
  }
  __syncthreads();

  for (int i = tid; i < 16 * H_SZ; i += 256) {
    int row = i >> 8, j = i & 255;
    float gi = (float)sG[row * G4H + j];
    float gf = (float)sG[row * G4H + 256 + j];
    float gg = (float)sG[row * G4H + 512 + j];
    float go = (float)sG[row * G4H + 768 + j];
    int gb = b0 + row;
    float cn = gf * c[(size_t)gb * H_SZ + j] + gi * gg;
    float hn = go * tanhf(cn);
    c[(size_t)gb * H_SZ + j] = cn;
    h[(size_t)gb * H_SZ + j] = hn;
    h16[(size_t)gb * H_SZ + j] = (_Float16)hn;
    if (hs16) hs16[((size_t)gb * TDEC + t) * H_SZ + j] = (_Float16)hn;
  }
}

// Conditioning: h += tanh(cond@W_ch^T + b_ch); c += tanh(cond@W_cc^T + b_cc)
__global__ void cond_kernel(const float* __restrict__ W_rhyme,
                            const int* __restrict__ rhyme_ids,
                            const float* __restrict__ syll,
                            const float* __restrict__ W_ch,
                            const float* __restrict__ b_ch,
                            const float* __restrict__ W_cc,
                            const float* __restrict__ b_cc,
                            float* __restrict__ h, float* __restrict__ c,
                            _Float16* __restrict__ h16) {
  int idx = blockIdx.x * blockDim.x + threadIdx.x;
  if (idx >= B_SZ * H_SZ) return;
  int b = idx / H_SZ, j = idx % H_SZ;
  int rid = rhyme_ids[b];
  const float* rv = W_rhyme + (size_t)rid * RE_SZ;
  float dh = b_ch[j], dc = b_cc[j];
  const float* wch = W_ch + (size_t)j * (RE_SZ + 1);
  const float* wcc = W_cc + (size_t)j * (RE_SZ + 1);
#pragma unroll 8
  for (int k = 0; k < RE_SZ; ++k) {
    float v = rv[k];
    dh += v * wch[k];
    dc += v * wcc[k];
  }
  float sy = syll[b];
  dh += sy * wch[RE_SZ];
  dc += sy * wcc[RE_SZ];
  float hn = h[idx] + tanhf(dh);
  float cn = c[idx] + tanhf(dc);
  h[idx] = hn;
  c[idx] = cn;
  h16[idx] = (_Float16)hn;
}

// ---------------------------------------------------------------------------
// token_logits = hs16(7424x256) @ W_out^T(256x32000) + b_out  (f32 out)
// Block: 16 rows x 256 cols, 4 waves x 4 n-tiles; K=256 (8 k-tiles).
// Prefetch next k-tile's B fragments (global_prefetch_b8) while computing.
// ---------------------------------------------------------------------------
__global__ __launch_bounds__(128) void gemm_out_kernel(
    const _Float16* __restrict__ hs16, const _Float16* __restrict__ WoutPk,
    const float* __restrict__ b_out, float* __restrict__ out) {
  __shared__ _Float16 sA[16 * H_SZ];
  int tid = threadIdx.x;
  int m0 = blockIdx.x * 16;
  const uint32_t* Ag = (const uint32_t*)(hs16 + (size_t)m0 * H_SZ);
  uint32_t* sAu = (uint32_t*)sA;
  for (int i = tid; i < 16 * H_SZ / 2; i += 128) sAu[i] = Ag[i];
  __syncthreads();

  int lane = tid & 31, w = tid >> 5;
  int hi = lane >> 4, ln = lane & 15;
  int ntBase = blockIdx.y * 16 + w * 4;

  v8f acc[4];
#pragma unroll
  for (int i = 0; i < 4; ++i)
#pragma unroll
    for (int r = 0; r < 8; ++r) acc[i][r] = 0.f;

#pragma unroll
  for (int kt = 0; kt < 8; ++kt) {
    v16h a = load_a_frag(sA, H_SZ, kt, lane);
#pragma unroll
    for (int i = 0; i < 4; ++i) {
      if (kt < 7) {
        // hide L2 latency of the next k-tile's fragment stream
        __builtin_prefetch(
            WoutPk + ((size_t)((ntBase + i) * 8 + kt + 1) * 32 + lane) * 16, 0,
            1);
      }
      v16h b = load_b_frag(WoutPk, ntBase + i, kt, 8, lane);
      acc[i] = wmma_f16(a, b, acc[i]);
    }
  }
#pragma unroll
  for (int i = 0; i < 4; ++i) {
    int col = (ntBase + i) * 16 + ln;
    float bb = b_out[col];
#pragma unroll
    for (int r = 0; r < 8; ++r) {
      int rg = m0 + r + 8 * hi;
      out[(size_t)rg * V_SZ + col] = acc[i][r] + bb;
    }
  }
}

// rhyme_logits = h_fin @ W_rc^T + b_rc  (small, VALU dot)
__global__ void rhyme_kernel(const float* __restrict__ h,
                             const float* __restrict__ W_rc,
                             const float* __restrict__ b_rc,
                             float* __restrict__ out) {
  int idx = blockIdx.x * blockDim.x + threadIdx.x;
  if (idx >= B_SZ * R_SZ) return;
  int b = idx / R_SZ, r = idx % R_SZ;
  const float* hb = h + (size_t)b * H_SZ;
  const float* wr = W_rc + (size_t)r * H_SZ;
  float s = b_rc[r];
#pragma unroll 8
  for (int k = 0; k < H_SZ; ++k) s += hb[k] * wr[k];
  out[idx] = s;
}

// ---------------------------------------------------------------------------
extern "C" void kernel_launch(void* const* d_in, const int* in_sizes, int n_in,
                              void* d_out, int out_size, void* d_ws,
                              size_t ws_size, hipStream_t stream) {
  const int*   prev_seq  = (const int*)d_in[0];
  const int*   next_seq  = (const int*)d_in[1];
  const float* syllables = (const float*)d_in[2];
  const int*   rhyme_ids = (const int*)d_in[3];
  const float* W_emb     = (const float*)d_in[4];
  const float* W_rhyme   = (const float*)d_in[5];
  const float* Wih_e     = (const float*)d_in[6];
  const float* Whh_e     = (const float*)d_in[7];
  const float* bih_e     = (const float*)d_in[8];
  const float* bhh_e     = (const float*)d_in[9];
  const float* Wih_d     = (const float*)d_in[10];
  const float* Whh_d     = (const float*)d_in[11];
  const float* bih_d     = (const float*)d_in[12];
  const float* bhh_d     = (const float*)d_in[13];
  const float* W_out     = (const float*)d_in[14];
  const float* b_out     = (const float*)d_in[15];
  const float* W_rc      = (const float*)d_in[16];
  const float* b_rc      = (const float*)d_in[17];
  const float* W_ch      = (const float*)d_in[18];
  const float* b_ch      = (const float*)d_in[19];
  const float* W_cc      = (const float*)d_in[20];
  const float* b_cc      = (const float*)d_in[21];

  float* out_tok   = (float*)d_out;
  float* out_rhyme = out_tok + (size_t)B_SZ * TDEC * V_SZ;

  // ---- workspace carve-out (256B aligned) ----
  size_t off = 0;
  char* base = (char*)d_ws;
  auto carve = [&](size_t bytes) -> char* {
    char* p = base + off;
    off += (bytes + 255) & ~(size_t)255;
    return p;
  };
  _Float16* Xe16   = (_Float16*)carve((size_t)B_SZ * S_SZ * E_SZ * 2);
  _Float16* Xd16   = (_Float16*)carve((size_t)B_SZ * TDEC * E_SZ * 2);
  _Float16* WihEpk = (_Float16*)carve((size_t)G4H * E_SZ * 2);
  _Float16* WihDpk = (_Float16*)carve((size_t)G4H * E_SZ * 2);
  _Float16* WhhEpk = (_Float16*)carve((size_t)G4H * H_SZ * 2);
  _Float16* WhhDpk = (_Float16*)carve((size_t)G4H * H_SZ * 2);
  _Float16* Woutpk = (_Float16*)carve((size_t)V_SZ * H_SZ * 2);
  float*    xprojE = (float*)carve((size_t)S_SZ * B_SZ * G4H * 4);
  float*    xprojD = (float*)carve((size_t)TDEC * B_SZ * G4H * 4);
  float*    hbuf   = (float*)carve((size_t)B_SZ * H_SZ * 4);
  float*    cbuf   = (float*)carve((size_t)B_SZ * H_SZ * 4);
  _Float16* h16    = (_Float16*)carve((size_t)B_SZ * H_SZ * 2);
  _Float16* hs16   = (_Float16*)carve((size_t)B_SZ * TDEC * H_SZ * 2);
  (void)ws_size; (void)n_in; (void)in_sizes; (void)out_size;

  // ---- pack weights into WMMA B-fragment layout ----
  auto packN = [&](const float* W, _Float16* dst, int Ng, int Kg) {
    long long tot = (long long)Ng * Kg;
    int blocks = (int)((tot + 255) / 256);
    hipLaunchKernelGGL(pack_b_kernel, dim3(blocks), dim3(256), 0, stream, W,
                       dst, Ng, Kg);
  };
  packN(Wih_e, WihEpk, G4H, E_SZ);
  packN(Wih_d, WihDpk, G4H, E_SZ);
  packN(Whh_e, WhhEpk, G4H, H_SZ);
  packN(Whh_d, WhhDpk, G4H, H_SZ);
  packN(W_out, Woutpk, V_SZ, H_SZ);

  // ---- gather embeddings (f16) ----
  {
    int rowsE = B_SZ * S_SZ;
    hipLaunchKernelGGL(gather_emb_kernel,
                       dim3((rowsE * E_SZ + 255) / 256), dim3(256), 0, stream,
                       W_emb, prev_seq, Xe16, rowsE, S_SZ, S_SZ);
    int rowsD = B_SZ * TDEC;
    hipLaunchKernelGGL(gather_emb_kernel,
                       dim3((rowsD * E_SZ + 255) / 256), dim3(256), 0, stream,
                       W_emb, next_seq, Xd16, rowsD, TDEC, S_SZ);
  }

  // ---- zero initial state ----
  hipLaunchKernelGGL(init_state_kernel, dim3((B_SZ * H_SZ + 255) / 256),
                     dim3(256), 0, stream, hbuf, cbuf, h16);

  // ---- input projections (time-independent, batched WMMA GEMM) ----
  hipLaunchKernelGGL(gemm_xproj_kernel, dim3(B_SZ * S_SZ / 16), dim3(256), 0,
                     stream, Xe16, WihEpk, bih_e, bhh_e, xprojE, S_SZ);
  hipLaunchKernelGGL(gemm_xproj_kernel, dim3(B_SZ * TDEC / 16), dim3(256), 0,
                     stream, Xd16, WihDpk, bih_d, bhh_d, xprojD, TDEC);

  // ---- encoder recurrence ----
  for (int t = 0; t < S_SZ; ++t) {
    hipLaunchKernelGGL(lstm_step_kernel, dim3(B_SZ / 16), dim3(256), 0, stream,
                       xprojE + (size_t)t * B_SZ * G4H, WhhEpk, h16, hbuf,
                       cbuf, (_Float16*)nullptr, 0);
  }

  // ---- conditioning of decoder initial state ----
  hipLaunchKernelGGL(cond_kernel, dim3((B_SZ * H_SZ + 255) / 256), dim3(256),
                     0, stream, W_rhyme, rhyme_ids, syllables, W_ch, b_ch,
                     W_cc, b_cc, hbuf, cbuf, h16);

  // ---- decoder recurrence (collect hs16) ----
  for (int t = 0; t < TDEC; ++t) {
    hipLaunchKernelGGL(lstm_step_kernel, dim3(B_SZ / 16), dim3(256), 0, stream,
                       xprojD + (size_t)t * B_SZ * G4H, WhhDpk, h16, hbuf,
                       cbuf, hs16, t);
  }

  // ---- token logits: the 121 GFLOP / 950 MB-output GEMM ----
  hipLaunchKernelGGL(gemm_out_kernel, dim3(B_SZ * TDEC / 16, V_SZ / 256),
                     dim3(128), 0, stream, hs16, Woutpk, b_out, out_tok);

  // ---- rhyme logits ----
  hipLaunchKernelGGL(rhyme_kernel, dim3((B_SZ * R_SZ + 255) / 256), dim3(256),
                     0, stream, hbuf, W_rc, b_rc, out_rhyme);
}
